// MultiHeadAttention_69020124446922
// MI455X (gfx1250) — matrix operand order repaired and checked by reading, not compiled, not run on hardware
//
#include <hip/hip_runtime.h>
#include <hip/hip_bf16.h>

#define EMB      1024
#define HEADS    16
#define HEAD_DIM 64
#define BATCH    4
#define SEQ      2048
#define BSROWS   (BATCH * SEQ)   // 8192

typedef __attribute__((ext_vector_type(16))) __bf16 v16bf;
typedef __attribute__((ext_vector_type(8)))  __bf16 v8bf;
typedef __attribute__((ext_vector_type(8)))  float  v8f;

// fp32 -> bf16 via native converter (v_cvt_pk_bf16_f32 on gfx1250)
__device__ __forceinline__ __bf16 f2bf(float f) { return (__bf16)f; }

// Low 32 bits of a generic pointer into __shared__ = wave-relative LDS byte offset.
__device__ __forceinline__ unsigned lds_off32(const void* p) {
    return (unsigned)(uintptr_t)p;
}

// CDNA5 async copy: global -> LDS, 16 bytes per lane, tracked by ASYNCcnt.
__device__ __forceinline__ void async_copy_b128(unsigned lds_dst, const void* gsrc) {
    asm volatile("global_load_async_to_lds_b128 %0, %1, off"
                 :: "v"(lds_dst), "v"(gsrc) : "memory");
}
// Wait until at most N async ops outstanding (async loads complete in order).
template <int N>
__device__ __forceinline__ void wait_async_le() {
    asm volatile("s_wait_asynccnt %0" :: "i"(N) : "memory");
}

__device__ __forceinline__ v8f wmma_bf16(v16bf a, v16bf b, v8f c) {
    // D = A(16x32 bf16) x B(32x16 bf16) + C(16x16 f32)
    return __builtin_amdgcn_wmma_f32_16x16x32_bf16(
        false, a, false, b, (short)0, c, false, false);
}

// A-fragment 16x32 bf16 from a row-major fp32 source.
// elements 0..7 -> k = half*8 + e ; elements 8..15 -> k = 16 + half*8 + (e-8)
__device__ __forceinline__ v16bf afrag_f32(const float* rowchunk, int half) {
    const float* p = rowchunk + half * 8;
    float4 f0 = *(const float4*)(p);
    float4 f1 = *(const float4*)(p + 4);
    float4 f2 = *(const float4*)(p + 16);
    float4 f3 = *(const float4*)(p + 20);
    v16bf a;
    a[0]  = f2bf(f0.x); a[1]  = f2bf(f0.y); a[2]  = f2bf(f0.z); a[3]  = f2bf(f0.w);
    a[4]  = f2bf(f1.x); a[5]  = f2bf(f1.y); a[6]  = f2bf(f1.z); a[7]  = f2bf(f1.w);
    a[8]  = f2bf(f2.x); a[9]  = f2bf(f2.y); a[10] = f2bf(f2.z); a[11] = f2bf(f2.w);
    a[12] = f2bf(f3.x); a[13] = f2bf(f3.y); a[14] = f2bf(f3.z); a[15] = f2bf(f3.w);
    return a;
}

// A-fragment 16x32 from a row-major bf16 source (global or LDS).
__device__ __forceinline__ v16bf afrag_bf16(const __bf16* rowchunk, int half) {
    const __bf16* p = rowchunk + half * 8;
    v8bf lo = *(const v8bf*)(p);
    v8bf hi = *(const v8bf*)(p + 16);
    v16bf a;
#pragma unroll
    for (int i = 0; i < 8; ++i) { a[i] = lo[i]; a[i + 8] = hi[i]; }
    return a;
}

// B-fragment 32x16: lane = column n; elements 0..7 -> k = half*8 + e ; elements 8..15 -> k = 16 + half*8 + (e-8)
__device__ __forceinline__ v16bf bfrag_bf16(const __bf16* colchunk, int half) {
    const __bf16* p = colchunk + half * 8;
    v8bf lo = *(const v8bf*)(p);
    v8bf hi = *(const v8bf*)(p + 16);
    v16bf b;
#pragma unroll
    for (int i = 0; i < 8; ++i) { b[i] = lo[i]; b[i + 8] = hi[i]; }
    return b;
}

// ---------------- Kernel 1: weight transpose + fp32->bf16 ----------------
__global__ void __launch_bounds__(256) wt_convert(const float* __restrict__ W,
                                                  __bf16* __restrict__ WT) {
    int idx = blockIdx.x * 256 + threadIdx.x;
    int k = idx >> 10;
    int n = idx & (EMB - 1);
    WT[(size_t)n * EMB + k] = f2bf(W[idx]);
}

// ---------------- Kernel 2: X[8192,1024] @ W + b -> head-split bf16 ----------------
// VMODE 0: out [B,H,S,D] (Q,K);  VMODE 1: out [B,H,D,S] (V transposed).
// 8 waves stacked in M; 64x64 B-tile double-buffered via async copy:
// one barrier pair covers 8 WMMAs per wave.
template <int VMODE>
__global__ void __launch_bounds__(256) qkv_gemm(const float*  __restrict__ X,
                                                const __bf16* __restrict__ WT,
                                                const float*  __restrict__ bias,
                                                __bf16*       __restrict__ out) {
    __shared__ __bf16 btile[2][64][64];   // 16KB double buffer, [n][k]

    const int tid  = threadIdx.x;
    const int lane = tid & 31;
    const int wave = tid >> 5;
    const int half = lane >> 4;
    const int c0   = lane & 15;
    const int m0   = blockIdx.x * 128 + wave * 16;
    const int n0   = blockIdx.y * 64;

    // staging: 64x64 tile = 512 x b128 chunks -> 2 chunks/thread
    const int ch0 = tid * 2, ch1 = tid * 2 + 1;
    const unsigned bd0 = lds_off32(&btile[0][ch0 >> 3][(ch0 & 7) * 8]);
    const unsigned bd1 = lds_off32(&btile[0][ch1 >> 3][(ch1 & 7) * 8]);
    const __bf16* bs0 = WT + (size_t)(n0 + (ch0 >> 3)) * EMB + (ch0 & 7) * 8;
    const __bf16* bs1 = WT + (size_t)(n0 + (ch1 >> 3)) * EMB + (ch1 & 7) * 8;

    v8f acc[4];
#pragma unroll
    for (int t = 0; t < 4; ++t) acc[t] = {};

    const float* arow = X + (size_t)(m0 + c0) * EMB;

    async_copy_b128(bd0, bs0);                          // prologue: stage buf 0
    async_copy_b128(bd1, bs1);
    for (int kk = 0; kk < EMB; kk += 64) {
        const int cur = (kk >> 6) & 1;
        const int nxt = cur ^ 1;
        __syncthreads();                                // buf[nxt] readers (prev iter) done
        const int kk2 = (kk + 64) & (EMB - 1);          // wrap: harmless redundant stage
        const unsigned bo = nxt * (unsigned)sizeof(btile[0]);
        async_copy_b128(bd0 + bo, bs0 + kk2);
        async_copy_b128(bd1 + bo, bs1 + kk2);
        __builtin_prefetch(arow + kk2, 0, 0);           // next A chunk -> global_prefetch
        v16bf a0 = afrag_f32(arow + kk, half);
        v16bf a1 = afrag_f32(arow + kk + 32, half);
        wait_async_le<2>();                             // buf[cur] (older batch) done
        __syncthreads();                                // ... for all waves
        v16bf bf[4];
#pragma unroll
        for (int t = 0; t < 4; ++t)
            bf[t] = bfrag_bf16(&btile[cur][t * 16 + c0][0], half);
#pragma unroll
        for (int t = 0; t < 4; ++t)
            acc[t] = wmma_bf16(a0, bf[t], acc[t]);
#pragma unroll
        for (int t = 0; t < 4; ++t)
            bf[t] = bfrag_bf16(&btile[cur][t * 16 + c0][32], half);
#pragma unroll
        for (int t = 0; t < 4; ++t)
            acc[t] = wmma_bf16(a1, bf[t], acc[t]);
    }

#pragma unroll
    for (int t = 0; t < 4; ++t) {
        const int c  = n0 + t * 16 + c0;
        const float bc = bias[c];
        const int h = c >> 6;
        const int d = c & 63;
#pragma unroll
        for (int r = 0; r < 8; ++r) {
            const int grow = m0 + r + 8 * half;     // C layout: row = r + 8*(lane>=16)
            const int bb = grow >> 11;              // / SEQ
            const int s  = grow & (SEQ - 1);
            const float v = acc[t][r] + bc;
            size_t addr;
            if (VMODE == 0) addr = (((size_t)bb * HEADS + h) * SEQ + s) * HEAD_DIM + d;
            else            addr = (((size_t)bb * HEADS + h) * HEAD_DIM + d) * SEQ + s;
            out[addr] = f2bf(v);
        }
    }
}

// ---------------- Kernel 3: flash-style attention ----------------
// grid = (SEQ/64, B*H), block = 128 (4 waves, 16 query rows each).
// K/V tiles shared by the 4 waves, double-buffered via async copy.
// Softmax runs in the exp2 domain: one fused scale (1/8 * log2 e), native v_exp.
__global__ void __launch_bounds__(128) attn_kernel(const __bf16* __restrict__ Q,
                                                   const __bf16* __restrict__ K,
                                                   const __bf16* __restrict__ Vt,
                                                   __bf16*       __restrict__ Ctx) {
    __shared__ __bf16 ktile[2][32][64];   // [key][d]   8KB double buffer
    __shared__ __bf16 vtile[2][64][32];   // [d][key]   8KB double buffer
    __shared__ __bf16 pbuf[4][16][32];    // per-wave P bounce, 4KB

    const int tid  = threadIdx.x;
    const int lane = tid & 31;
    const int wave = tid >> 5;
    const int half = lane >> 4;
    const int c0   = lane & 15;
    const int bh   = blockIdx.y;
    const int b    = bh >> 4;
    const int h    = bh & 15;
    const int q0   = blockIdx.x * 64 + wave * 16;

    const __bf16* Qb = Q  + ((size_t)bh * SEQ + q0) * HEAD_DIM;
    const __bf16* Kb = K  + (size_t)bh * SEQ * HEAD_DIM;
    const __bf16* Vb = Vt + (size_t)bh * HEAD_DIM * SEQ;

    // staging: 256 x b128 per tile -> 2 chunks/thread/tile
    const int kc0 = tid * 2, kc1 = tid * 2 + 1;
    const unsigned kd0 = lds_off32(&ktile[0][kc0 >> 3][(kc0 & 7) * 8]);
    const unsigned kd1 = lds_off32(&ktile[0][kc1 >> 3][(kc1 & 7) * 8]);
    const unsigned vd0 = lds_off32(&vtile[0][kc0 >> 2][(kc0 & 3) * 8]);
    const unsigned vd1 = lds_off32(&vtile[0][kc1 >> 2][(kc1 & 3) * 8]);
    const __bf16* ksrc0 = Kb + (size_t)(kc0 >> 3) * HEAD_DIM + (kc0 & 7) * 8;
    const __bf16* ksrc1 = Kb + (size_t)(kc1 >> 3) * HEAD_DIM + (kc1 & 7) * 8;
    const __bf16* vsrc0 = Vb + (size_t)(kc0 >> 2) * SEQ + (kc0 & 3) * 8;
    const __bf16* vsrc1 = Vb + (size_t)(kc1 >> 2) * SEQ + (kc1 & 3) * 8;

    auto stage = [&](int buf, int kbase) {
        const unsigned ko = buf * (unsigned)sizeof(ktile[0]);
        const unsigned vo = buf * (unsigned)sizeof(vtile[0]);
        async_copy_b128(kd0 + ko, ksrc0 + (size_t)kbase * HEAD_DIM);
        async_copy_b128(kd1 + ko, ksrc1 + (size_t)kbase * HEAD_DIM);
        async_copy_b128(vd0 + vo, vsrc0 + kbase);
        async_copy_b128(vd1 + vo, vsrc1 + kbase);
    };

    // Q A-fragments (16 queries x 64d = 2 x K32), loaded once from global
    const __bf16* qrow = Qb + (size_t)c0 * HEAD_DIM;
    const v16bf aq0 = afrag_bf16(qrow, half);
    const v16bf aq1 = afrag_bf16(qrow + 32, half);

    v8f o[4];
#pragma unroll
    for (int t = 0; t < 4; ++t) o[t] = {};
    float m_r[8], l_r[8];
#pragma unroll
    for (int r = 0; r < 8; ++r) { m_r[r] = -1e30f; l_r[r] = 0.f; }

    // 1/sqrt(64) * log2(e): softmax carried in the exp2 domain
    const float cscale = 0.125f * 1.44269504088896340736f;

    stage(0, 0);                                        // prologue
    for (int kb = 0; kb < SEQ; kb += 32) {
        const int cur = (kb >> 5) & 1;
        const int nxt = cur ^ 1;
        __syncthreads();                                // buf[nxt] readers done
        const int kb2 = (kb + 32) & (SEQ - 1);          // wrap: harmless
        stage(nxt, kb2);
        wait_async_le<4>();                             // cur batch (4 older ops) done
        __syncthreads();

        // scores: two 16x16 f32 fragments over 32 keys, K-dim 64 (2 WMMAs each)
        v16bf bk00 = bfrag_bf16(&ktile[cur][c0][0], half);
        v16bf bk01 = bfrag_bf16(&ktile[cur][c0][32], half);
        v16bf bk10 = bfrag_bf16(&ktile[cur][16 + c0][0], half);
        v16bf bk11 = bfrag_bf16(&ktile[cur][16 + c0][32], half);
        v8f s0 = {}, s1 = {};
        s0 = wmma_bf16(aq0, bk00, s0);
        s0 = wmma_bf16(aq1, bk01, s0);
        s1 = wmma_bf16(aq0, bk10, s1);
        s1 = wmma_bf16(aq1, bk11, s1);

        // online softmax, base 2 (row = r + 8*half; reduce across the 16-lane half)
        float p0[8], p1[8], alpha[8];
#pragma unroll
        for (int r = 0; r < 8; ++r) {
            float x0 = s0[r] * cscale, x1 = s1[r] * cscale;
            float bm = fmaxf(x0, x1);
#pragma unroll
            for (int off = 1; off < 16; off <<= 1)
                bm = fmaxf(bm, __shfl_xor(bm, off, 32));
            float nm = fmaxf(m_r[r], bm);
            alpha[r] = exp2f(m_r[r] - nm);
            p0[r] = exp2f(x0 - nm);
            p1[r] = exp2f(x1 - nm);
            float rs = p0[r] + p1[r];
#pragma unroll
            for (int off = 1; off < 16; off <<= 1)
                rs += __shfl_xor(rs, off, 32);
            l_r[r] = l_r[r] * alpha[r] + rs;
            m_r[r] = nm;
        }
#pragma unroll
        for (int t = 0; t < 4; ++t)
#pragma unroll
            for (int r = 0; r < 8; ++r) o[t][r] *= alpha[r];

        // P (C-layout) -> LDS -> reload in A-fragment layout (per-wave, in-order DS)
#pragma unroll
        for (int r = 0; r < 8; ++r) {
            const int row = r + 8 * half;
            pbuf[wave][row][c0]      = f2bf(p0[r]);
            pbuf[wave][row][c0 + 16] = f2bf(p1[r]);
        }
        asm volatile("s_wait_dscnt 0" ::: "memory");
        const v16bf ap = afrag_bf16(&pbuf[wave][c0][0], half);

        // O += P(16x32) @ V(32x64)
        v16bf bv[4];
#pragma unroll
        for (int t = 0; t < 4; ++t)
            bv[t] = bfrag_bf16(&vtile[cur][t * 16 + c0][0], half);
#pragma unroll
        for (int t = 0; t < 4; ++t)
            o[t] = wmma_bf16(ap, bv[t], o[t]);
    }

    // normalize and write context [B, S, E] in bf16
#pragma unroll
    for (int t = 0; t < 4; ++t) {
        const int col = h * HEAD_DIM + t * 16 + c0;
#pragma unroll
        for (int r = 0; r < 8; ++r) {
            const int row = q0 + r + 8 * half;
            const float v = o[t][r] / l_r[r];
            Ctx[((size_t)b * SEQ + row) * EMB + col] = f2bf(v);
        }
    }
}

// ---------------- Kernel 4: Ctx[8192,1024] @ Wo + bo -> f32 output ----------------
__global__ void __launch_bounds__(256) out_gemm(const __bf16* __restrict__ Ctx,
                                                const __bf16* __restrict__ WT,
                                                const float*  __restrict__ bias,
                                                float*        __restrict__ out) {
    __shared__ __bf16 btile[2][64][64];   // 16KB double buffer

    const int tid  = threadIdx.x;
    const int lane = tid & 31;
    const int wave = tid >> 5;
    const int half = lane >> 4;
    const int c0   = lane & 15;
    const int m0   = blockIdx.x * 128 + wave * 16;
    const int n0   = blockIdx.y * 64;

    const int ch0 = tid * 2, ch1 = tid * 2 + 1;
    const unsigned bd0 = lds_off32(&btile[0][ch0 >> 3][(ch0 & 7) * 8]);
    const unsigned bd1 = lds_off32(&btile[0][ch1 >> 3][(ch1 & 7) * 8]);
    const __bf16* bs0 = WT + (size_t)(n0 + (ch0 >> 3)) * EMB + (ch0 & 7) * 8;
    const __bf16* bs1 = WT + (size_t)(n0 + (ch1 >> 3)) * EMB + (ch1 & 7) * 8;

    v8f acc[4];
#pragma unroll
    for (int t = 0; t < 4; ++t) acc[t] = {};

    const __bf16* arow = Ctx + (size_t)(m0 + c0) * EMB;

    async_copy_b128(bd0, bs0);
    async_copy_b128(bd1, bs1);
    for (int kk = 0; kk < EMB; kk += 64) {
        const int cur = (kk >> 6) & 1;
        const int nxt = cur ^ 1;
        __syncthreads();
        const int kk2 = (kk + 64) & (EMB - 1);
        const unsigned bo = nxt * (unsigned)sizeof(btile[0]);
        async_copy_b128(bd0 + bo, bs0 + kk2);
        async_copy_b128(bd1 + bo, bs1 + kk2);
        __builtin_prefetch(arow + kk2, 0, 0);
        v16bf a0 = afrag_bf16(arow + kk, half);
        v16bf a1 = afrag_bf16(arow + kk + 32, half);
        wait_async_le<2>();
        __syncthreads();
        v16bf bf[4];
#pragma unroll
        for (int t = 0; t < 4; ++t)
            bf[t] = bfrag_bf16(&btile[cur][t * 16 + c0][0], half);
#pragma unroll
        for (int t = 0; t < 4; ++t)
            acc[t] = wmma_bf16(a0, bf[t], acc[t]);
#pragma unroll
        for (int t = 0; t < 4; ++t)
            bf[t] = bfrag_bf16(&btile[cur][t * 16 + c0][32], half);
#pragma unroll
        for (int t = 0; t < 4; ++t)
            acc[t] = wmma_bf16(a1, bf[t], acc[t]);
    }

#pragma unroll
    for (int t = 0; t < 4; ++t) {
        const int c = n0 + t * 16 + c0;
        const float bc = bias[c];
#pragma unroll
        for (int r = 0; r < 8; ++r) {
            const int grow = m0 + r + 8 * half;
            out[(size_t)grow * EMB + c] = acc[t][r] + bc;
        }
    }
}

extern "C" void kernel_launch(void* const* d_in, const int* in_sizes, int n_in,
                              void* d_out, int out_size, void* d_ws, size_t ws_size,
                              hipStream_t stream) {
    const float* query = (const float*)d_in[0];
    const float* key   = (const float*)d_in[1];
    const float* value = (const float*)d_in[2];
    const float* Wq    = (const float*)d_in[3];
    const float* bq    = (const float*)d_in[4];
    const float* Wk    = (const float*)d_in[5];
    const float* bk    = (const float*)d_in[6];
    const float* Wv    = (const float*)d_in[7];
    const float* bv    = (const float*)d_in[8];
    const float* Wo    = (const float*)d_in[9];
    const float* bo    = (const float*)d_in[10];

    char* ws = (char*)d_ws;
    const size_t WBYTES = (size_t)EMB * EMB * sizeof(__bf16);                       // 2 MB
    const size_t QBYTES = (size_t)BATCH * HEADS * SEQ * HEAD_DIM * sizeof(__bf16);  // 16 MB
    __bf16* WqT = (__bf16*)(ws);
    __bf16* WkT = (__bf16*)(ws + 1 * WBYTES);
    __bf16* WvT = (__bf16*)(ws + 2 * WBYTES);
    __bf16* WoT = (__bf16*)(ws + 3 * WBYTES);
    __bf16* Qw  = (__bf16*)(ws + 4 * WBYTES);
    __bf16* Kw  = (__bf16*)(ws + 4 * WBYTES + 1 * QBYTES);
    __bf16* Vw  = (__bf16*)(ws + 4 * WBYTES + 2 * QBYTES);   // [B,H,D,S]
    __bf16* Cw  = (__bf16*)(ws + 4 * WBYTES + 3 * QBYTES);   // [B,S,E]

    const int wtBlocks = (EMB * EMB) / 256;
    wt_convert<<<wtBlocks, 256, 0, stream>>>(Wq, WqT);
    wt_convert<<<wtBlocks, 256, 0, stream>>>(Wk, WkT);
    wt_convert<<<wtBlocks, 256, 0, stream>>>(Wv, WvT);
    wt_convert<<<wtBlocks, 256, 0, stream>>>(Wo, WoT);

    dim3 gp(BSROWS / 128, EMB / 64);       // (64, 16)
    qkv_gemm<0><<<gp, 256, 0, stream>>>(query, WqT, bq, Qw);
    qkv_gemm<0><<<gp, 256, 0, stream>>>(key,   WkT, bk, Kw);
    qkv_gemm<1><<<gp, 256, 0, stream>>>(value, WvT, bv, Vw);

    dim3 ga(SEQ / 64, BATCH * HEADS);      // (32, 64)
    attn_kernel<<<ga, 128, 0, stream>>>(Qw, Kw, Vw, Cw);

    out_gemm<<<gp, 256, 0, stream>>>(Cw, WoT, bo, (float*)d_out);
}
